// Encoder_33277406609766
// MI455X (gfx1250) — compile-verified, matrix operand order checked by reading, not run on hardware
//
#include <hip/hip_runtime.h>
#include <hip/hip_bf16.h>

// ---------------------------------------------------------------------------
// Types for CDNA5 WMMA (wave32, 16x16x32 bf16 -> f32)
// ---------------------------------------------------------------------------
typedef __bf16 bf16_t;
typedef bf16_t v16bf __attribute__((ext_vector_type(16)));
typedef float  v8f   __attribute__((ext_vector_type(8)));

union FragU { bf16_t b[16]; v16bf v; };

__device__ __forceinline__ v8f wmma_bf16(v16bf a, v16bf b, v8f c) {
    // D = A(16x32 bf16) * B(32x16 bf16) + C(16x16 f32)
    return __builtin_amdgcn_wmma_f32_16x16x32_bf16(
        /*neg_a=*/false, a, /*neg_b=*/false, b,
        /*c_mod=*/(short)0, c, /*reuse_a=*/false, /*reuse_b=*/false);
}

// XOR-butterfly within a 16-lane DPP row, pure VALU (no LDS round trip).
__device__ __forceinline__ float dpp_xor16(float x, const int which) {
    int xi = __builtin_bit_cast(int, x);
    int r;
    switch (which) {  // dpp_ctrl = ROW_XMASK(0x160) | mask ; must be ICE
        case 1:  r = __builtin_amdgcn_update_dpp(xi, xi, 0x161, 0xf, 0xf, true); break;
        case 2:  r = __builtin_amdgcn_update_dpp(xi, xi, 0x162, 0xf, 0xf, true); break;
        case 4:  r = __builtin_amdgcn_update_dpp(xi, xi, 0x164, 0xf, 0xf, true); break;
        default: r = __builtin_amdgcn_update_dpp(xi, xi, 0x168, 0xf, 0xf, true); break;
    }
    return __builtin_bit_cast(float, r);
}
__device__ __forceinline__ float rowmax16(float v) {
    v = fmaxf(v, dpp_xor16(v, 1));
    v = fmaxf(v, dpp_xor16(v, 2));
    v = fmaxf(v, dpp_xor16(v, 4));
    v = fmaxf(v, dpp_xor16(v, 8));
    return v;
}
__device__ __forceinline__ float rowsum16(float v) {
    v += dpp_xor16(v, 1);
    v += dpp_xor16(v, 2);
    v += dpp_xor16(v, 4);
    v += dpp_xor16(v, 8);
    return v;
}

// Fragment gather from LDS stored as [outer][stride] bf16 (outer = M-row for A
// or N-col for B^T-staged tiles). Per CDNA5 ISA 16-bit layout:
//   lane: outer = outer0 + (lane&15), khalf = lane>>4
//   element e: k = (e&7) + 8*khalf + 16*(e>>3)
__device__ __forceinline__ v16bf load_frag(const bf16_t* base, int outer0,
                                           int koff, int stride, int lane) {
    FragU f;
    const bf16_t* p = base + (size_t)(outer0 + (lane & 15)) * stride
                           + koff + 8 * (lane >> 4);
#pragma unroll
    for (int e = 0; e < 16; ++e) f.b[e] = p[(e & 7) + 16 * (e >> 3)];
    return f.v;
}

// B-fragment gather from LDS stored K-major: [k][stride] bf16; lane owns column
// n0 + (lane&15), element e -> row k = (e&7) + 8*(lane>>4) + 16*(e>>3).
__device__ __forceinline__ v16bf load_fragT(const bf16_t* base, int n0,
                                            int stride, int lane) {
    FragU f;
    const bf16_t* p = base + n0 + (lane & 15) + (size_t)(8 * (lane >> 4)) * stride;
#pragma unroll
    for (int e = 0; e < 16; ++e) f.b[e] = p[(size_t)((e & 7) + 16 * (e >> 3)) * stride];
    return f.v;
}

// ---------------------------------------------------------------------------
// GEMM: C[M,N] = A[M,K] @ W[K,N] + bias  (+ReLU | +residual)
// fp32 in/out, bf16 WMMA internally. Tile 128x128, K-step 32, 256 threads
// (8 waves, each wave -> 32x64 output = 2x4 WMMA tiles).
// EPI: 0 = bias, 1 = bias+ReLU, 2 = bias+residual
// ---------------------------------------------------------------------------
template <int EPI>
__global__ __launch_bounds__(256)
void gemm_bf16_wmma(const float* __restrict__ A, const float* __restrict__ W,
                    const float* __restrict__ bias, const float* __restrict__ R,
                    float* __restrict__ C, int M, int N, int K) {
    __shared__ bf16_t As[128][34];   // [m][k] bf16, padded (odd-bank stride)
    __shared__ bf16_t Bs[128][34];   // [n][k] bf16 (transposed while loading)

    const int tid  = threadIdx.x;
    const int lane = tid & 31;
    const int wave = tid >> 5;
    const int bm = blockIdx.y * 128;
    const int bn = blockIdx.x * 128;
    const int wm = (wave >> 1) * 32;   // 0,32,64,96
    const int wn = (wave & 1) * 64;    // 0,64

    v8f acc[2][4];
#pragma unroll
    for (int mi = 0; mi < 2; ++mi)
#pragma unroll
        for (int ni = 0; ni < 4; ++ni)
#pragma unroll
            for (int i = 0; i < 8; ++i) acc[mi][ni][i] = 0.0f;

    const int ar = tid >> 1, akc = (tid & 1) * 16;     // A: row, k-offset (16 floats)
    const int bkr = tid >> 3, bnc = (tid & 7) * 16;    // W: k-row, n-offset (16 floats)

    for (int k0 = 0; k0 < K; k0 += 32) {
        const float* srcA = A + (size_t)(bm + ar) * K + k0 + akc;
        const float* srcB = W + (size_t)(k0 + bkr) * N + bn + bnc;
#pragma unroll
        for (int e = 0; e < 16; ++e) As[ar][akc + e] = (bf16_t)srcA[e];
#pragma unroll
        for (int e = 0; e < 16; ++e) Bs[bnc + e][bkr] = (bf16_t)srcB[e];
        if (k0 + 32 < K) {                       // hint next K-tile into cache
            __builtin_prefetch(srcA + 32, 0, 1); // -> global_prefetch_b8
            __builtin_prefetch(srcB + (size_t)32 * N, 0, 1);
        }
        __syncthreads();

        v16bf af[2], bfr[4];
#pragma unroll
        for (int mi = 0; mi < 2; ++mi) af[mi]  = load_frag(&As[0][0], wm + mi * 16, 0, 34, lane);
#pragma unroll
        for (int ni = 0; ni < 4; ++ni) bfr[ni] = load_frag(&Bs[0][0], wn + ni * 16, 0, 34, lane);
#pragma unroll
        for (int mi = 0; mi < 2; ++mi)
#pragma unroll
            for (int ni = 0; ni < 4; ++ni)
                acc[mi][ni] = wmma_bf16(af[mi], bfr[ni], acc[mi][ni]);
        __syncthreads();
    }

    // Epilogue: C-frag element r -> row = r + 8*(lane>>4), col = lane&15
    const int kh = lane >> 4, ln = lane & 15;
#pragma unroll
    for (int mi = 0; mi < 2; ++mi)
#pragma unroll
        for (int ni = 0; ni < 4; ++ni)
#pragma unroll
            for (int r = 0; r < 8; ++r) {
                int row = bm + wm + mi * 16 + r + 8 * kh;
                int col = bn + wn + ni * 16 + ln;
                float v = acc[mi][ni][r] + bias[col];
                if (EPI == 1) v = fmaxf(v, 0.0f);
                if (EPI == 2) v += R[(size_t)row * N + col];
                C[(size_t)row * N + col] = v;
            }
}

// ---------------------------------------------------------------------------
// Flash attention: one block per (b,h); 8 waves x 32 query rows = S=256.
// scores = Q K^T / 32, online softmax, O = P V.  DH=64, key blocks of 32.
// ---------------------------------------------------------------------------
__global__ __launch_bounds__(256)
void attn_kernel(const float* __restrict__ qkv, float* __restrict__ out) {
    const int bh = blockIdx.x, b = bh >> 4, h = bh & 15;
    const int tid = threadIdx.x, lane = tid & 31, wave = tid >> 5;
    const int q0 = wave * 32;
    const int kh = lane >> 4, ln = lane & 15;

    __shared__ bf16_t Ks[32][66];        // [key][dim] bf16
    __shared__ bf16_t Vs[32][66];        // [key][dim] bf16
    __shared__ bf16_t Ps[8][32][34];     // per-wave P staging [m][k]

    const size_t base = (size_t)b * 256 * 3072;  // qkv: [B][S][3*E], E=1024

    // Q fragments straight from global (fp32 -> bf16)
    v16bf qf[2][2];
#pragma unroll
    for (int mi = 0; mi < 2; ++mi)
#pragma unroll
        for (int ks = 0; ks < 2; ++ks) {
            FragU f;
            const float* qrow = qkv + base + (size_t)(q0 + mi * 16 + ln) * 3072
                                + h * 64 + ks * 32 + 8 * kh;
#pragma unroll
            for (int e = 0; e < 16; ++e) f.b[e] = (bf16_t)qrow[(e & 7) + 16 * (e >> 3)];
            qf[mi][ks] = f.v;
        }

    float mrun[16], lsum[16];
#pragma unroll
    for (int i = 0; i < 16; ++i) { mrun[i] = -1e30f; lsum[i] = 0.0f; }
    v8f of[2][4];
#pragma unroll
    for (int mi = 0; mi < 2; ++mi)
#pragma unroll
        for (int ni = 0; ni < 4; ++ni)
#pragma unroll
            for (int i = 0; i < 8; ++i) of[mi][ni][i] = 0.0f;

    const float inv_scale = 1.0f / 32.0f;        // 1/sqrt(E)

    for (int j = 0; j < 256; j += 32) {
        __syncthreads();
        for (int i = tid; i < 2048; i += 256) {  // 32 keys x 64 dims
            int key = i >> 6, d = i & 63;
            size_t idx = base + (size_t)(j + key) * 3072 + h * 64 + d;
            Ks[key][d] = (bf16_t)qkv[idx + 1024];
            Vs[key][d] = (bf16_t)qkv[idx + 2048];
        }
        __syncthreads();

        // S-tile = Q @ K^T : A = Q frag, B lane-col = key, k = dim
        v8f sc[2][2];
#pragma unroll
        for (int mi = 0; mi < 2; ++mi)
#pragma unroll
            for (int ni = 0; ni < 2; ++ni)
#pragma unroll
                for (int i = 0; i < 8; ++i) sc[mi][ni][i] = 0.0f;
        v16bf kb[2][2];
#pragma unroll
        for (int ni = 0; ni < 2; ++ni)
#pragma unroll
            for (int ks = 0; ks < 2; ++ks)
                kb[ni][ks] = load_frag(&Ks[0][0], ni * 16, ks * 32, 66, lane);
#pragma unroll
        for (int mi = 0; mi < 2; ++mi)
#pragma unroll
            for (int ni = 0; ni < 2; ++ni)
#pragma unroll
                for (int ks = 0; ks < 2; ++ks)
                    sc[mi][ni] = wmma_bf16(qf[mi][ks], kb[ni][ks], sc[mi][ni]);

        // Online softmax; row reductions via DPP row_xmask (VALU only)
#pragma unroll
        for (int mi = 0; mi < 2; ++mi)
#pragma unroll
            for (int r = 0; r < 8; ++r) {
                float a = sc[mi][0][r] * inv_scale;
                float c = sc[mi][1][r] * inv_scale;
                float mx = rowmax16(fmaxf(a, c));
                int id = mi * 8 + r;
                float mnew  = fmaxf(mrun[id], mx);
                float alpha = __expf(mrun[id] - mnew);
                float p0 = __expf(a - mnew), p1 = __expf(c - mnew);
                sc[mi][0][r] = p0; sc[mi][1][r] = p1;
                float rs = rowsum16(p0 + p1);
                lsum[id] = lsum[id] * alpha + rs;
                mrun[id] = mnew;
#pragma unroll
                for (int ni = 0; ni < 4; ++ni) of[mi][ni][r] *= alpha;
            }

        // Restage P (C-layout -> A-layout) through per-wave LDS
#pragma unroll
        for (int mi = 0; mi < 2; ++mi)
#pragma unroll
            for (int nc = 0; nc < 2; ++nc)
#pragma unroll
                for (int r = 0; r < 8; ++r)
                    Ps[wave][mi * 16 + r + 8 * kh][nc * 16 + ln] = (bf16_t)sc[mi][nc][r];
        asm volatile("s_wait_dscnt 0" ::: "memory");

        v16bf pf[2], vb[4];
#pragma unroll
        for (int mi = 0; mi < 2; ++mi)
            pf[mi] = load_frag(&Ps[wave][0][0], mi * 16, 0, 34, lane);
#pragma unroll
        for (int ni = 0; ni < 4; ++ni)
            vb[ni] = load_fragT(&Vs[0][0], ni * 16, 66, lane);   // lane col = dim, k = key
#pragma unroll
        for (int mi = 0; mi < 2; ++mi)
#pragma unroll
            for (int ni = 0; ni < 4; ++ni)
                of[mi][ni] = wmma_bf16(pf[mi], vb[ni], of[mi][ni]);
    }

    // O = O / l ; scatter back into [B][S][E] head slice
#pragma unroll
    for (int mi = 0; mi < 2; ++mi)
#pragma unroll
        for (int ni = 0; ni < 4; ++ni)
#pragma unroll
            for (int r = 0; r < 8; ++r) {
                int row = q0 + mi * 16 + r + 8 * kh;
                int col = ni * 16 + ln;
                out[((size_t)(b * 256 + row)) * 1024 + h * 64 + col] =
                    of[mi][ni][r] * (1.0f / lsum[mi * 8 + r]);
            }
}

// ---------------------------------------------------------------------------
// LayerNorm over E=1024 per row; 256 threads/row
// ---------------------------------------------------------------------------
__global__ __launch_bounds__(256)
void layernorm_kernel(const float* __restrict__ in, const float* __restrict__ gamma,
                      const float* __restrict__ beta, float* __restrict__ out) {
    const int row = blockIdx.x, tid = threadIdx.x;
    const float* x = in + (size_t)row * 1024;
    __shared__ float red[8];

    float v[4], s = 0.0f;
#pragma unroll
    for (int e = 0; e < 4; ++e) { v[e] = x[e * 256 + tid]; s += v[e]; }
    s = rowsum16(s);
    s += __shfl_xor(s, 16, 32);                 // cross-row step
    if ((tid & 31) == 0) red[tid >> 5] = s;
    __syncthreads();
    float tot = 0.0f;
#pragma unroll
    for (int i = 0; i < 8; ++i) tot += red[i];
    const float mu = tot * (1.0f / 1024.0f);
    __syncthreads();

    float sq = 0.0f;
#pragma unroll
    for (int e = 0; e < 4; ++e) { float d = v[e] - mu; sq += d * d; }
    sq = rowsum16(sq);
    sq += __shfl_xor(sq, 16, 32);
    if ((tid & 31) == 0) red[tid >> 5] = sq;
    __syncthreads();
    float vt = 0.0f;
#pragma unroll
    for (int i = 0; i < 8; ++i) vt += red[i];
    const float rstd = rsqrtf(vt * (1.0f / 1024.0f) + 1e-5f);

    float* o = out + (size_t)row * 1024;
#pragma unroll
    for (int e = 0; e < 4; ++e) {
        int c = e * 256 + tid;
        o[c] = (v[e] - mu) * rstd * gamma[c] + beta[c];
    }
}

// ---------------------------------------------------------------------------
// x += sinusoidal PE(s, e)   (x: [B][S][E])
// ---------------------------------------------------------------------------
__global__ __launch_bounds__(256)
void add_pe_kernel(float* __restrict__ x) {
    size_t idx = (size_t)blockIdx.x * 256 + threadIdx.x;
    int e = (int)(idx & 1023);
    int s = (int)((idx >> 10) & 255);
    int j2 = e & ~1;                               // arange(0,dim,2) value
    float div = __expf(-9.210340371976184f * (float)j2 * (1.0f / 1024.0f));
    float ang = (float)s * div;
    x[idx] += (e & 1) ? __cosf(ang) : __sinf(ang);
}

// ---------------------------------------------------------------------------
// Host-side orchestration
// ---------------------------------------------------------------------------
extern "C" void kernel_launch(void* const* d_in, const int* in_sizes, int n_in,
                              void* d_out, int out_size, void* d_ws, size_t ws_size,
                              hipStream_t stream) {
    const float* patches = (const float*)d_in[0];
    const float* W_emb   = (const float*)d_in[1];
    const float* b_emb   = (const float*)d_in[2];
    const float* qkv_w   = (const float*)d_in[3];
    const float* qkv_b   = (const float*)d_in[4];
    const float* proj_w  = (const float*)d_in[5];
    const float* proj_b  = (const float*)d_in[6];
    const float* ff1_w   = (const float*)d_in[7];
    const float* ff1_b   = (const float*)d_in[8];
    const float* ff2_w   = (const float*)d_in[9];
    const float* ff2_b   = (const float*)d_in[10];
    const float* gamma   = (const float*)d_in[11];
    const float* beta    = (const float*)d_in[12];

    const int M = 32 * 256;   // 8192 tokens
    float* ws   = (float*)d_ws;
    float* x    = ws;                              // M*1024
    float* qkv  = x    + (size_t)M * 1024;         // M*3072
    float* att  = qkv  + (size_t)M * 3072;         // M*1024
    float* t    = att  + (size_t)M * 1024;         // M*1024
    float* y    = t    + (size_t)M * 1024;         // M*1024
    float* hbuf = y    + (size_t)M * 1024;         // M*4096

    dim3 blk(256);

    // Embed: x = patches @ W_emb + b_emb ; then += PE
    gemm_bf16_wmma<0><<<dim3(1024 / 128, M / 128), blk, 0, stream>>>(
        patches, W_emb, b_emb, nullptr, x, M, 1024, 768);
    add_pe_kernel<<<(M * 1024) / 256, blk, 0, stream>>>(x);

    for (int l = 0; l < 12; ++l) {
        gemm_bf16_wmma<0><<<dim3(3072 / 128, M / 128), blk, 0, stream>>>(
            x, qkv_w + (size_t)l * 1024 * 3072, qkv_b + (size_t)l * 3072,
            nullptr, qkv, M, 3072, 1024);
        attn_kernel<<<512, blk, 0, stream>>>(qkv, att);
        gemm_bf16_wmma<2><<<dim3(1024 / 128, M / 128), blk, 0, stream>>>(
            att, proj_w + (size_t)l * 1024 * 1024, proj_b + (size_t)l * 1024,
            x, t, M, 1024, 1024);
        layernorm_kernel<<<M, blk, 0, stream>>>(t, gamma + (size_t)l * 1024,
                                                beta + (size_t)l * 1024, y);
        gemm_bf16_wmma<1><<<dim3(4096 / 128, M / 128), blk, 0, stream>>>(
            y, ff1_w + (size_t)l * 1024 * 4096, ff1_b + (size_t)l * 4096,
            nullptr, hbuf, M, 4096, 1024);
        float* dst = (l == 11) ? (float*)d_out : x;   // scan carry = ff
        gemm_bf16_wmma<0><<<dim3(1024 / 128, M / 128), blk, 0, stream>>>(
            hbuf, ff2_w + (size_t)l * 4096 * 1024, ff2_b + (size_t)l * 1024,
            nullptr, dst, M, 1024, 4096);
    }
}
